// MambaLM_4767413699114
// MI455X (gfx1250) — compile-verified, matrix operand order checked by reading, not run on hardware
//
#include <hip/hip_runtime.h>

#define DEV __device__ __forceinline__

typedef __bf16 bf16_t;
typedef __attribute__((ext_vector_type(8)))  __bf16 v8bf;
typedef __attribute__((ext_vector_type(16))) __bf16 v16bf;
typedef __attribute__((ext_vector_type(8)))  float  v8f;

// ---------------- model constants ----------------
constexpr int BB  = 2;
constexpr int LL  = 1024;
constexpr int DM  = 512;           // d_model
constexpr int DI  = 1024;          // d_inner
constexpr int DS  = 16;            // d_state
constexpr int DC  = 4;             // d_conv
constexpr int DTR = 32;            // dt_rank
constexpr int NL  = 6;
constexpr int VOC = 32000;
constexpr int MR  = BB * LL;       // 2048 token rows

// ---------------- helpers ----------------
DEV unsigned short f2bf(float f) {
  unsigned u = __float_as_uint(f);
  return (unsigned short)((u + 0x7FFFu + ((u >> 16) & 1u)) >> 16);
}

// Load one 16x32 bf16 fragment (A- or B-matrix layout per CDNA5 ISA 7.12.2):
// lanes 0-15 : row = lane,    elems 0-7 = K k0..k0+7,   elems 8-15 = K k0+16..k0+23
// lanes 16-31: row = lane-16, elems 0-7 = K k0+8..k0+15, elems 8-15 = K k0+24..k0+31
DEV v16bf load_frag16x32(const bf16_t* __restrict__ base, int ld, int lane, int k0) {
  const int m  = lane & 15;
  const int hi = lane >> 4;
  const bf16_t* p = base + (size_t)m * ld + k0 + hi * 8;
  v8bf lo  = *(const v8bf*)(p);
  v8bf hi8 = *(const v8bf*)(p + 16);
  return __builtin_shufflevector(lo, hi8, 0,1,2,3,4,5,6,7,8,9,10,11,12,13,14,15);
}

// ---------------- kernels ----------------
__global__ void embed_k(const int* __restrict__ tok, const float* __restrict__ tab,
                        float* __restrict__ x) {
  const int row = blockIdx.x;
  const int t = tok[row];
  for (int c = threadIdx.x; c < DM; c += blockDim.x)
    x[(size_t)row * DM + c] = tab[(size_t)t * DM + c];
}

__global__ void cvt_bf16_k(const float* __restrict__ in, unsigned short* __restrict__ out, int n) {
  const int i = blockIdx.x * blockDim.x + threadIdx.x;
  if (i < n) out[i] = f2bf(in[i]);
}

// C[M,N] = A[M,K](bf16) * W[N,K](bf16)^T  (+ optional f32 residual), f32 out.
// Block = 256 threads = 8 waves; wave tile = 32x32 (2x2 WMMA frags); block tile = 64M x 128N.
__global__ __launch_bounds__(256) void wmma_gemm_k(
    const bf16_t* __restrict__ A, int lda,
    const bf16_t* __restrict__ W, int ldb,
    float* __restrict__ C, int ldc,
    const float* __restrict__ addsrc,
    int N, int K)
{
  const int lane = threadIdx.x & 31;
  const int wave = threadIdx.x >> 5;   // 0..7
  const int wm = wave >> 2;            // 0..1
  const int wn = wave & 3;             // 0..3
  const int m_base = blockIdx.y * 64 + wm * 32;
  const int n_base = blockIdx.x * 128 + wn * 32;
  if (n_base >= N) return;             // wave-uniform: EXEC stays all-ones for WMMA

  v8f acc00 = {}, acc01 = {}, acc10 = {}, acc11 = {};
  const bf16_t* A0 = A + (size_t)m_base * lda;
  const bf16_t* A1 = A0 + (size_t)16 * lda;
  const bf16_t* B0 = W + (size_t)n_base * ldb;
  const bf16_t* B1 = B0 + (size_t)16 * ldb;

  for (int k0 = 0; k0 < K; k0 += 32) {
    v16bf a0 = load_frag16x32(A0, lda, lane, k0);
    v16bf a1 = load_frag16x32(A1, lda, lane, k0);
    v16bf b0 = load_frag16x32(B0, ldb, lane, k0);
    v16bf b1 = load_frag16x32(B1, ldb, lane, k0);
    if (k0 + 32 < K) {  // uniform; lowers to global_prefetch_b8
      __builtin_prefetch(A0 + (size_t)(lane & 15) * lda + k0 + 32, 0, 1);
      __builtin_prefetch(B0 + (size_t)(lane & 15) * ldb + k0 + 32, 0, 1);
    }
    acc00 = __builtin_amdgcn_wmma_f32_16x16x32_bf16(false, a0, false, b0, (short)0, acc00, false, false);
    acc01 = __builtin_amdgcn_wmma_f32_16x16x32_bf16(false, a0, false, b1, (short)0, acc01, false, false);
    acc10 = __builtin_amdgcn_wmma_f32_16x16x32_bf16(false, a1, false, b0, (short)0, acc10, false, false);
    acc11 = __builtin_amdgcn_wmma_f32_16x16x32_bf16(false, a1, false, b1, (short)0, acc11, false, false);
  }

  // C/D layout: VGPR v -> row = v + 8*(lane>>4), col = lane&15
  const int col = lane & 15;
  const int hi  = lane >> 4;
  const v8f accs[4] = {acc00, acc01, acc10, acc11};
#pragma unroll
  for (int ti = 0; ti < 2; ++ti) {
#pragma unroll
    for (int tj = 0; tj < 2; ++tj) {
      const v8f a = accs[ti * 2 + tj];
#pragma unroll
      for (int v = 0; v < 8; ++v) {
        const int r = m_base + ti * 16 + 8 * hi + v;
        const int c = n_base + tj * 16 + col;
        float o = a[v];
        if (addsrc) o += addsrc[(size_t)r * ldc + c];
        C[(size_t)r * ldc + c] = o;
      }
    }
  }
}

// causal depthwise conv (width 4) + bias + SiLU over xi = first half of xz
__global__ void conv_silu_k(const float* __restrict__ xz, const float* __restrict__ cw,
                            const float* __restrict__ cb, float* __restrict__ xc) {
  const int idx = blockIdx.x * blockDim.x + threadIdx.x;
  if (idx >= MR * DI) return;
  const int d = idx & (DI - 1);
  const int r = idx >> 10;            // b*L + l
  const int l = r & (LL - 1);
  float acc = cb[d];
#pragma unroll
  for (int j = 0; j < DC; ++j) {
    const int ll = l + j - (DC - 1);
    if (ll >= 0)
      acc += cw[d * DC + j] * xz[(size_t)(r + j - (DC - 1)) * (2 * DI) + d];
  }
  xc[idx] = acc / (1.f + __expf(-acc));     // SiLU
}

// fused selective scan: one lane per (b,d,n); 16-lane shuffle reduce for y = C^T h
__global__ void scan_k(const float* __restrict__ dpre, const float* __restrict__ dt_bias,
                       const float* __restrict__ xc,   const float* __restrict__ dbc,
                       const float* __restrict__ a_log, const float* __restrict__ dskip,
                       const float* __restrict__ xz,   float* __restrict__ yg) {
  const int gid = blockIdx.x * blockDim.x + threadIdx.x;
  const int n  = gid & (DS - 1);
  const int pd = gid >> 4;
  if (pd >= BB * DI) return;
  const int d = pd & (DI - 1);
  const int b = pd >> 10;
  const float Ad   = -__expf(a_log[d * DS + n]);   // A = -exp(A_log)
  const float Dd   = dskip[d];
  const float bias = dt_bias[d];
  float h = 0.f;
  for (int l = 0; l < LL; ++l) {
    const int r = b * LL + l;
    const float dp = dpre[(size_t)r * DI + d] + bias;
    const float delta = (dp > 20.f) ? dp : log1pf(__expf(dp));   // softplus
    const float x  = xc[(size_t)r * DI + d];
    const float Bn = dbc[r * 64 + DTR + n];
    const float Cn = dbc[r * 64 + DTR + DS + n];
    h = __expf(delta * Ad) * h + (delta * x) * Bn;
    float y = h * Cn;
    y += __shfl_xor(y, 8, 16);
    y += __shfl_xor(y, 4, 16);
    y += __shfl_xor(y, 2, 16);
    y += __shfl_xor(y, 1, 16);
    if (n == 0) {
      const float z = xz[(size_t)r * (2 * DI) + DI + d];
      yg[(size_t)r * DI + d] = (y + x * Dd) * (z / (1.f + __expf(-z)));
    }
  }
}

// LayerNorm over 512 features, emit bf16 for the head GEMM
__global__ void layernorm_bf16_k(const float* __restrict__ x, const float* __restrict__ w,
                                 const float* __restrict__ b, unsigned short* __restrict__ out) {
  __shared__ float red[256];
  const int row = blockIdx.x;
  const int t = threadIdx.x;
  const float* xr = x + (size_t)row * DM;
  float s = 0.f;
  for (int c = t; c < DM; c += 256) s += xr[c];
  red[t] = s; __syncthreads();
  for (int o = 128; o > 0; o >>= 1) { if (t < o) red[t] += red[t + o]; __syncthreads(); }
  const float mu = red[0] / DM;
  __syncthreads();
  float v = 0.f;
  for (int c = t; c < DM; c += 256) { const float dd = xr[c] - mu; v += dd * dd; }
  red[t] = v; __syncthreads();
  for (int o = 128; o > 0; o >>= 1) { if (t < o) red[t] += red[t + o]; __syncthreads(); }
  const float inv = rsqrtf(red[0] / DM + 1e-5f);
  for (int c = t; c < DM; c += 256)
    out[(size_t)row * DM + c] = f2bf((xr[c] - mu) * inv * w[c] + b[c]);
}

// ---------------- host launch ----------------
extern "C" void kernel_launch(void* const* d_in, const int* in_sizes, int n_in,
                              void* d_out, int out_size, void* d_ws, size_t ws_size,
                              hipStream_t stream) {
  (void)in_sizes; (void)n_in; (void)out_size; (void)ws_size;
  const int*   tokens = (const int*)d_in[0];
  const float* embed  = (const float*)d_in[1];
  const float* w_in   = (const float*)d_in[2];
  const float* w_conv = (const float*)d_in[3];
  const float* b_conv = (const float*)d_in[4];
  const float* w_xp   = (const float*)d_in[5];
  const float* w_dt   = (const float*)d_in[6];
  const float* b_dt   = (const float*)d_in[7];
  const float* a_log  = (const float*)d_in[8];
  const float* d_skip = (const float*)d_in[9];
  const float* w_out  = (const float*)d_in[10];
  const float* nw     = (const float*)d_in[11];
  const float* nb     = (const float*)d_in[12];
  const float* w_head = (const float*)d_in[13];
  float* logits = (float*)d_out;

  char* ws = (char*)d_ws;
  size_t off = 0;
  auto take = [&](size_t bytes) -> char* {
    char* p = ws + off;
    off += (bytes + 255) & ~(size_t)255;
    return p;
  };
  float* X    = (float*)take((size_t)MR * DM * 4);
  float* XZ   = (float*)take((size_t)MR * 2 * DI * 4);
  float* XC   = (float*)take((size_t)MR * DI * 4);
  float* DBC  = (float*)take((size_t)MR * 64 * 4);
  float* DPRE = (float*)take((size_t)MR * DI * 4);
  float* YG   = (float*)take((size_t)MR * DI * 4);
  unsigned short* Xbf    = (unsigned short*)take((size_t)MR * DM * 2);
  unsigned short* XCbf   = (unsigned short*)take((size_t)MR * DI * 2);
  unsigned short* DBCbf  = (unsigned short*)take((size_t)MR * 64 * 2);
  unsigned short* YGbf   = (unsigned short*)take((size_t)MR * DI * 2);
  unsigned short* XNbf   = (unsigned short*)take((size_t)MR * DM * 2);
  unsigned short* Winbf  = (unsigned short*)take((size_t)2 * DI * DM * 2);
  unsigned short* Wxpbf  = (unsigned short*)take((size_t)(DTR + 2 * DS) * DI * 2);
  unsigned short* Wdtbf  = (unsigned short*)take((size_t)DI * DTR * 2);
  unsigned short* Woutbf = (unsigned short*)take((size_t)DM * DI * 2);
  unsigned short* Whdbf  = (unsigned short*)take((size_t)VOC * DM * 2);

  auto cvt = [&](const float* src, unsigned short* dst, int n) {
    cvt_bf16_k<<<(n + 255) / 256, 256, 0, stream>>>(src, dst, n);
  };
  auto gemm = [&](const unsigned short* A, int lda, const unsigned short* W, int ldb,
                  float* C, int ldc, const float* addsrc, int N, int K) {
    dim3 grid((N + 127) / 128, MR / 64);
    wmma_gemm_k<<<grid, 256, 0, stream>>>((const bf16_t*)A, lda, (const bf16_t*)W, ldb,
                                          C, ldc, addsrc, N, K);
  };

  // embedding + one-time head weight conversion
  embed_k<<<MR, 256, 0, stream>>>(tokens, embed, X);
  cvt(w_head, Whdbf, VOC * DM);

  for (int i = 0; i < NL; ++i) {
    // in_proj: XZ[2048,2048] = X * Win^T
    cvt(X, Xbf, MR * DM);
    cvt(w_in + (size_t)i * 2 * DI * DM, Winbf, 2 * DI * DM);
    gemm(Xbf, DM, Winbf, DM, XZ, 2 * DI, nullptr, 2 * DI, DM);

    // causal depthwise conv + SiLU
    conv_silu_k<<<(MR * DI) / 256, 256, 0, stream>>>(
        XZ, w_conv + (size_t)i * DI * DC, b_conv + (size_t)i * DI, XC);

    // x_proj: DBC[2048,64] = XC * Wxp^T
    cvt(XC, XCbf, MR * DI);
    cvt(w_xp + (size_t)i * (DTR + 2 * DS) * DI, Wxpbf, (DTR + 2 * DS) * DI);
    gemm(XCbf, DI, Wxpbf, DI, DBC, 64, nullptr, 64, DI);

    // dt_proj: DPRE[2048,1024] = DBC[:, :32] * Wdt^T   (A slice via lda=64)
    cvt(DBC, DBCbf, MR * 64);
    cvt(w_dt + (size_t)i * DI * DTR, Wdtbf, DI * DTR);
    gemm(DBCbf, 64, Wdtbf, DTR, DPRE, DI, nullptr, DI, DTR);

    // fused softplus + selective scan + D-skip + SiLU(z) gate
    scan_k<<<(BB * DI * DS) / 256, 256, 0, stream>>>(
        DPRE, b_dt + (size_t)i * DI, XC, DBC,
        a_log + (size_t)i * DI * DS, d_skip + (size_t)i * DI, XZ, YG);

    // out_proj + residual (in-place on X: each element read+written by same thread)
    cvt(YG, YGbf, MR * DI);
    cvt(w_out + (size_t)i * DM * DI, Woutbf, DM * DI);
    gemm(YGbf, DI, Woutbf, DI, X, DM, X, DM, DI);
  }

  // final LayerNorm -> bf16, then head GEMM to f32 logits
  layernorm_bf16_k<<<MR, 256, 0, stream>>>(X, nw, nb, XNbf);
  gemm(XNbf, DM, Whdbf, DM, logits, VOC, nullptr, VOC, DM);
}
